// StrokePainter_59777354826096
// MI455X (gfx1250) — matrix operand order repaired and checked by reading, not compiled, not run on hardware
//
#include <hip/hip_runtime.h>

typedef __attribute__((ext_vector_type(16))) _Float16 v16h;
typedef __attribute__((ext_vector_type(8)))  float    v8f;

#define LN_EPS 1e-5f

__device__ __forceinline__ int imin_(int a, int b) { return a < b ? a : b; }

// ---------------------------------------------------------------------------
// WMMA tile loaders (f32 memory -> f16 registers, CDNA5 16x16x32 layouts)
// ---------------------------------------------------------------------------
// A operand: 16(M) x 32(K).  Lane L: row = L&15.  Lanes 0-15 hold K 0-15,
// lanes 16-31 hold K 16-31, contiguous within the lane (matches B loaders).
__device__ __forceinline__ v16h ld_a_tile(const float* src, int ld, int row_base, int row_max, int k0)
{
  int lane = threadIdx.x & 31;
  int r  = imin_(row_base + (lane & 15), row_max);
  int kk = k0 + ((lane >> 4) << 4);
  const float* p = src + (size_t)r * ld + kk;
  v16h out;
#pragma unroll
  for (int j = 0; j < 8; ++j) {
    out[j]     = (_Float16)p[j];
    out[j + 8] = (_Float16)p[j + 8];
  }
  return out;
}

// B operand: 32(K) x 16(N), fed from Bt stored row-major [N,K] (i.e. weights
// W[n][k]).  Lane L: col n = L&15; lanes 0-15 hold K=0..15, lanes 16-31 hold
// K=16..31, contiguous within the lane (same k order as ld_a_tile).
__device__ __forceinline__ v16h ld_b_tile_nk(const float* src, int ld, int col_base, int col_max, int k0)
{
  int lane = threadIdx.x & 31;
  int n  = imin_(col_base + (lane & 15), col_max);
  int kk = k0 + ((lane >> 4) << 4);
  const float* p = src + (size_t)n * ld + kk;
  v16h out;
#pragma unroll
  for (int j = 0; j < 16; ++j) out[j] = (_Float16)p[j];
  return out;
}

// Same B operand, but fed from a matrix stored row-major [K,N] (e.g. V).
__device__ __forceinline__ v16h ld_b_tile_kn(const float* src, int ld, int col_base, int col_max, int k0)
{
  int lane = threadIdx.x & 31;
  int n  = imin_(col_base + (lane & 15), col_max);
  int kk = k0 + ((lane >> 4) << 4);
  const float* p = src + (size_t)kk * ld + n;
  v16h out;
#pragma unroll
  for (int j = 0; j < 16; ++j) out[j] = (_Float16)p[(size_t)j * ld];
  return out;
}

// C/D 16x16 f32: lane L: n = L&15; vgpr r: m = r + 8*(L>>4).
__device__ __forceinline__ void st_tile(v8f acc, int m_tile, int n_tile,
    float* C, int ldc, const float* bias, const float* res, int ldr,
    int M, int N, float alpha, int act)
{
  int lane = threadIdx.x & 31;
  int n  = n_tile + (lane & 15);
  int m0 = m_tile + ((lane >> 4) << 3);
  if (n >= N) return;
  float bv = bias ? bias[n] : 0.0f;
#pragma unroll
  for (int r = 0; r < 8; ++r) {
    int m = m0 + r;
    if (m < M) {
      float v = alpha * acc[r] + bv;
      if (res) v += res[(size_t)m * ldr + n];
      if (act == 1) v = fmaxf(v, 0.0f);
      C[(size_t)m * ldc + n] = v;
    }
  }
}

// ---------------------------------------------------------------------------
// Generic batched GEMM: C[m,n] = alpha * sum_k A[m,k] * B[k,n] (+bias+res, act)
// A: row-major [M,K]. B: if B_KXN, row-major [K,N]; else Bt row-major [N,K].
// blockIdx.z -> (batch, head) via nh; per-operand (batch, head) strides.
// 256 threads = 8 waves; wave computes a 32x32 tile; block tile 128x64.
// ---------------------------------------------------------------------------
template<bool B_KXN>
__global__ void __launch_bounds__(256)
gemm_kernel(const float* __restrict__ A, int lda, long long sAb, long long sAh,
            const float* __restrict__ Bt, int ldb, long long sBb, long long sBh,
            const float* __restrict__ bias,
            const float* __restrict__ res, long long sRb, long long sRh, int ldr,
            float* __restrict__ C, int ldc, long long sCb, long long sCh,
            int M, int N, int K, int nh, float alpha, int act)
{
  int z  = blockIdx.z;
  int bb = z / nh, hh = z - bb * nh;
  A  += (long long)bb * sAb + (long long)hh * sAh;
  Bt += (long long)bb * sBb + (long long)hh * sBh;
  C  += (long long)bb * sCb + (long long)hh * sCh;
  if (res) res += (long long)bb * sRb + (long long)hh * sRh;

  int wave = threadIdx.x >> 5;
  int m0 = blockIdx.y * 128 + (wave & 3) * 32;
  int n0 = blockIdx.x * 64  + (wave >> 2) * 32;
  int mmax = M - 1, nmax = N - 1;

#if defined(__gfx1250__)
  // speculative warm-up prefetch of this block's first A / B tile rows
  __builtin_prefetch(A + (size_t)imin_(m0, mmax) * lda, 0, 3);
  __builtin_prefetch(Bt + (B_KXN ? (size_t)imin_(n0, nmax)
                                 : (size_t)imin_(n0, nmax) * ldb), 0, 3);
#endif

  v8f acc00 = {}, acc01 = {}, acc10 = {}, acc11 = {};
  for (int k0 = 0; k0 < K; k0 += 32) {
    v16h a0 = ld_a_tile(A, lda, m0,      mmax, k0);
    v16h a1 = ld_a_tile(A, lda, m0 + 16, mmax, k0);
    v16h b0, b1;
    if (B_KXN) {
      b0 = ld_b_tile_kn(Bt, ldb, n0,      nmax, k0);
      b1 = ld_b_tile_kn(Bt, ldb, n0 + 16, nmax, k0);
    } else {
      b0 = ld_b_tile_nk(Bt, ldb, n0,      nmax, k0);
      b1 = ld_b_tile_nk(Bt, ldb, n0 + 16, nmax, k0);
    }
    acc00 = __builtin_amdgcn_wmma_f32_16x16x32_f16(false, a0, false, b0, (short)0, acc00, false, false);
    acc01 = __builtin_amdgcn_wmma_f32_16x16x32_f16(false, a0, false, b1, (short)0, acc01, false, false);
    acc10 = __builtin_amdgcn_wmma_f32_16x16x32_f16(false, a1, false, b0, (short)0, acc10, false, false);
    acc11 = __builtin_amdgcn_wmma_f32_16x16x32_f16(false, a1, false, b1, (short)0, acc11, false, false);
  }
  st_tile(acc00, m0,      n0,      C, ldc, bias, res, ldr, M, N, alpha, act);
  st_tile(acc01, m0,      n0 + 16, C, ldc, bias, res, ldr, M, N, alpha, act);
  st_tile(acc10, m0 + 16, n0,      C, ldc, bias, res, ldr, M, N, alpha, act);
  st_tile(acc11, m0 + 16, n0 + 16, C, ldc, bias, res, ldr, M, N, alpha, act);
}

// ---------------------------------------------------------------------------
// CNN kernels
// ---------------------------------------------------------------------------
__global__ void conv3x3_kernel(const float* __restrict__ in, const float* __restrict__ w,
                               const float* __restrict__ bias, float* __restrict__ out,
                               int Bn, int Ci, int Co, int Hin, int Win,
                               int stride, int Hout, int Wout)
{
  int idx = blockIdx.x * blockDim.x + threadIdx.x;
  int total = Bn * Co * Hout * Wout;
  if (idx >= total) return;
  int wo = idx % Wout;
  int t  = idx / Wout;
  int ho = t % Hout; t /= Hout;
  int co = t % Co;
  int b  = t / Co;
  int hy[3], wx[3];
#pragma unroll
  for (int d = 0; d < 3; ++d) {           // reflect pad 1
    int h = ho * stride - 1 + d;
    hy[d] = h < 0 ? -h : (h >= Hin ? 2 * Hin - 2 - h : h);
    int x = wo * stride - 1 + d;
    wx[d] = x < 0 ? -x : (x >= Win ? 2 * Win - 2 - x : x);
  }
  float acc = bias[co];
  const float* wp = w + (size_t)co * Ci * 9;
  const float* ip = in + (size_t)b * Ci * Hin * Win;
  for (int ci = 0; ci < Ci; ++ci) {
    const float* ipc = ip + (size_t)ci * Hin * Win;
    const float* wpc = wp + ci * 9;
#pragma unroll
    for (int r = 0; r < 3; ++r)
#pragma unroll
      for (int c = 0; c < 3; ++c)
        acc += wpc[r * 3 + c] * ipc[hy[r] * Win + wx[c]];
  }
  out[idx] = acc;
}

__global__ void bn_stats_kernel(const float* __restrict__ x, float* __restrict__ mean,
                                float* __restrict__ var, int Bn, int C, int HW)
{
  __shared__ float rs[256], rq[256];
  int c = blockIdx.x, t = threadIdx.x;
  int n = Bn * HW;
  float s = 0.f, q = 0.f;
  for (int i = t; i < n; i += 256) {
    int b = i / HW, p = i - b * HW;
    float v = x[((size_t)(b * C + c)) * HW + p];
    s += v; q += v * v;
  }
  rs[t] = s; rq[t] = q; __syncthreads();
  for (int s2 = 128; s2 > 0; s2 >>= 1) {
    if (t < s2) { rs[t] += rs[t + s2]; rq[t] += rq[t + s2]; }
    __syncthreads();
  }
  if (t == 0) {
    float m = rs[0] / (float)n;
    mean[c] = m;
    var[c]  = rq[0] / (float)n - m * m;
  }
}

__global__ void bn_apply_relu_kernel(float* __restrict__ x, const float* __restrict__ g,
                                     const float* __restrict__ beta, const float* __restrict__ mean,
                                     const float* __restrict__ var, int total, int C, int HW)
{
  int i = blockIdx.x * blockDim.x + threadIdx.x;
  if (i >= total) return;
  int c = (i / HW) % C;
  float v = g[c] * (x[i] - mean[c]) * rsqrtf(var[c] + LN_EPS) + beta[c];
  x[i] = fmaxf(v, 0.f);
}

// tok[b, t, c] : c<128 from img feat, else canvas feat; feat layout (4,128,1024)
__global__ void build_tokens_kernel(const float* __restrict__ a, const float* __restrict__ b,
                                    float* __restrict__ tok)
{
  int i = blockIdx.x * 256 + threadIdx.x;   // 4*1024*256 = 2^20
  if (i >= 4 * 1024 * 256) return;
  int c  = i & 255;
  int t  = (i >> 8) & 1023;
  int bz = i >> 18;
  const float* srcp = (c < 128) ? a : b;
  int cc = c & 127;
  tok[i] = srcp[(((size_t)bz * 128 + cc) << 10) + t];
}

// ---------------------------------------------------------------------------
// Transformer pointwise kernels
// ---------------------------------------------------------------------------
__global__ void layernorm_kernel(const float* __restrict__ x, const float* __restrict__ g,
                                 const float* __restrict__ b, float* __restrict__ y)
{
  __shared__ float rs[256], rq[256];
  int row = blockIdx.x, t = threadIdx.x;
  float v = x[(size_t)row * 256 + t];
  rs[t] = v; rq[t] = v * v; __syncthreads();
  for (int s2 = 128; s2 > 0; s2 >>= 1) {
    if (t < s2) { rs[t] += rs[t + s2]; rq[t] += rq[t + s2]; }
    __syncthreads();
  }
  float m   = rs[0] * (1.f / 256.f);
  float var = rq[0] * (1.f / 256.f) - m * m;
  y[(size_t)row * 256 + t] = g[t] * (v - m) * rsqrtf(var + LN_EPS) + b[t];
}

__global__ void softmax_rows_kernel(float* __restrict__ x, int L)
{
  __shared__ float red[256];
  long long row = blockIdx.x;
  float* xr = x + row * (long long)L;
  int t = threadIdx.x;
  float mx = -3.0e38f;
  for (int i = t; i < L; i += 256) mx = fmaxf(mx, xr[i]);
  red[t] = mx; __syncthreads();
  for (int s = 128; s > 0; s >>= 1) { if (t < s) red[t] = fmaxf(red[t], red[t + s]); __syncthreads(); }
  mx = red[0]; __syncthreads();
  float sum = 0.f;
  for (int i = t; i < L; i += 256) { float e = expf(xr[i] - mx); xr[i] = e; sum += e; }
  red[t] = sum; __syncthreads();
  for (int s = 128; s > 0; s >>= 1) { if (t < s) red[t] += red[t + s]; __syncthreads(); }
  float inv = 1.f / red[0];
  for (int i = t; i < L; i += 256) xr[i] *= inv;
}

__global__ void bcast_kernel(const float* __restrict__ src, float* __restrict__ dst, int total, int period)
{
  int i = blockIdx.x * blockDim.x + threadIdx.x;
  if (i < total) dst[i] = src[i % period];
}

// ---------------------------------------------------------------------------
// Heads + rasterizer
// ---------------------------------------------------------------------------
__device__ __forceinline__ float sigmoidf_(float v) { return 1.0f / (1.0f + expf(-v)); }

// Per-stroke record (208 floats): [0..199] bezier pts (x,y interleaved),
// [200..203] culling bbox (xmin,xmax,ymin,ymax incl. underflow radius),
// [204] denom, [205..207] rgb color.
__global__ void heads_kernel(const float* __restrict__ sp2, float* __restrict__ recs)
{
  int bs = blockIdx.x * blockDim.x + threadIdx.x;   // (b*128 + s), 512 total
  if (bs >= 512) return;
  const float* r = sp2 + (size_t)bs * 16;
  float p0x = sigmoidf_(r[0]), p0y = sigmoidf_(r[1]);
  float p1x = sigmoidf_(r[2]), p1y = sigmoidf_(r[3]);
  float p2x = sigmoidf_(r[4]), p2y = sigmoidf_(r[5]);
  float c0  = sigmoidf_(r[6]), c1 = sigmoidf_(r[7]), c2 = sigmoidf_(r[8]);
  float brush = sigmoidf_(r[9]) * 20.0f;
  float denom = 0.005f * brush * brush;
  float* rec = recs + (size_t)bs * 208;
  float minx = 3.0e38f, maxx = -3.0e38f, miny = 3.0e38f, maxy = -3.0e38f;
  for (int p = 0; p < 100; ++p) {
    float t  = (float)p * (1.0f / 99.0f);
    float u  = 1.0f - t;
    float w0 = u * u, w1 = 2.0f * u * t, w2 = t * t;
    float px = (w0 * p0x + w1 * p1x + w2 * p2x) * 128.0f;
    float py = (w0 * p0y + w1 * p1y + w2 * p2y) * 128.0f;
    rec[2 * p + 0] = px;
    rec[2 * p + 1] = py;
    minx = fminf(minx, px); maxx = fmaxf(maxx, px);
    miny = fminf(miny, py); maxy = fmaxf(maxy, py);
  }
  // exp(-d2/denom) underflows f32 (< min normal) once d2 > ~88*denom; such
  // contributions cannot perturb a normal-range accumulator.
  float rcut = sqrtf(88.0f * denom);
  rec[200] = minx - rcut;
  rec[201] = maxx + rcut;
  rec[202] = miny - rcut;
  rec[203] = maxy + rcut;
  rec[204] = denom;
  rec[205] = c0; rec[206] = c1; rec[207] = c2;
}

// max_p exp(-d2/denom) == exp(-min_p d2 / denom): one exp per (pixel, stroke).
// Stroke records staged into LDS with CDNA5 async-to-LDS DMA (ASYNCcnt).
__global__ void rasterize_kernel(const float* __restrict__ recs, float* __restrict__ out)
{
  __shared__ float rec[208];
  int b = blockIdx.y;
  int t = threadIdx.x;
  int pix = blockIdx.x * 256 + t;           // 0..16383
  float fx = (float)(pix >> 7);             // X == row index
  float fy = (float)(pix & 127);            // Y == col index
  float a0 = 0.f, a1 = 0.f, a2 = 0.f;
  for (int s = 0; s < 128; ++s) {
    const float* src = recs + (size_t)(b * 128 + s) * 208;
#if defined(__gfx1250__)
    if (t < 208) {
      unsigned lds_off = (unsigned)(size_t)(void*)&rec[t];
      const float* gp = src + t;
      asm volatile("global_load_async_to_lds_b32 %0, %1, off"
                   :: "v"(lds_off), "v"(gp) : "memory");
    }
    asm volatile("s_wait_asynccnt 0x0" ::: "memory");
#else
    if (t < 208) rec[t] = src[t];
#endif
    __syncthreads();
    if (fx >= rec[200] && fx <= rec[201] && fy >= rec[202] && fy <= rec[203]) {
      float mind2 = 3.0e38f;
      for (int p = 0; p < 100; ++p) {
        float dx = fx - rec[2 * p];
        float dy = fy - rec[2 * p + 1];
        float d2 = dx * dx + dy * dy;
        mind2 = fminf(mind2, d2);
      }
      float mask = expf(-mind2 / rec[204]);
      a0 += fminf(fmaxf(rec[205] * mask, 0.f), 1.f);
      a1 += fminf(fmaxf(rec[206] * mask, 0.f), 1.f);
      a2 += fminf(fmaxf(rec[207] * mask, 0.f), 1.f);
    }
    __syncthreads();   // protect LDS before next stroke's async writes
  }
  size_t base = (size_t)b * 3 * 16384 + pix;
  out[base]             = a0;
  out[base + 16384]     = a1;
  out[base + 2 * 16384] = a2;
}

// ---------------------------------------------------------------------------
// Host side
// ---------------------------------------------------------------------------
namespace {
struct AttnP { const float *bk, *bo, *bq, *bv, *wk, *wo, *wq, *wv; };
struct FFP   { const float *b1, *b2, *w1, *w2; };
struct LNP   { const float *b, *g; };
struct ConvP { const float *b, *beta, *g, *w; };
} // namespace

static void linear(hipStream_t st, const float* A, const float* W, const float* bias,
                   const float* res, float* C, int M, int N, int K, int act)
{
  dim3 grid((N + 63) / 64, (M + 127) / 128, 1);
  gemm_kernel<false><<<grid, 256, 0, st>>>(A, K, 0, 0, W, K, 0, 0, bias,
                                           res, 0, 0, N, C, N, 0, 0,
                                           M, N, K, 1, 1.0f, act);
}

static void run_mha(hipStream_t st, const float* xq, int Lq, const float* xkv, int Lk,
                    const AttnP& p, const float* resid,
                    float* qb, float* kb, float* vb, float* scores, float* attnout, float* outbuf)
{
  const int B = 4, NH = 8;
  linear(st, xq,  p.wq, p.bq, nullptr, qb, B * Lq, 256, 256, 0);
  linear(st, xkv, p.wk, p.bk, nullptr, kb, B * Lk, 256, 256, 0);
  linear(st, xkv, p.wv, p.bv, nullptr, vb, B * Lk, 256, 256, 0);
  { // scores[b,h,q,k] = (1/sqrt(32)) * Q_h K_h^T
    dim3 grid((Lk + 63) / 64, (Lq + 127) / 128, B * NH);
    gemm_kernel<false><<<grid, 256, 0, st>>>(
        qb, 256, (long long)Lq * 256, 32,
        kb, 256, (long long)Lk * 256, 32,
        nullptr, nullptr, 0, 0, 0,
        scores, Lk, (long long)NH * Lq * Lk, (long long)Lq * Lk,
        Lq, Lk, 32, NH, 0.17677669529663687f, 0);
  }
  softmax_rows_kernel<<<B * NH * Lq, 256, 0, st>>>(scores, Lk);
  { // attnout[b,q,h*32+d] = P V  (B operand read from [K,N]-layout V head view)
    dim3 grid(1, (Lq + 127) / 128, B * NH);
    gemm_kernel<true><<<grid, 256, 0, st>>>(
        scores, Lk, (long long)NH * Lq * Lk, (long long)Lq * Lk,
        vb, 256, (long long)Lk * 256, 32,
        nullptr, nullptr, 0, 0, 0,
        attnout, 256, (long long)Lq * 256, 32,
        Lq, 32, Lk, NH, 1.0f, 0);
  }
  linear(st, attnout, p.wo, p.bo, resid, outbuf, B * Lq, 256, 256, 0);
}

extern "C" void kernel_launch(void* const* d_in, const int* in_sizes, int n_in,
                              void* d_out, int out_size, void* d_ws, size_t ws_size,
                              hipStream_t stream)
{
  (void)in_sizes; (void)n_in; (void)out_size; (void)ws_size;
  int idx = 0;
  auto nxt = [&]() { return (const float*)d_in[idx++]; };

  // --- JAX pytree (alphabetical dict key) leaf order ---
  const float* img    = nxt();
  const float* canvas = nxt();
  const float* c1b = nxt();           // conv1x1.b
  const float* c1w = nxt();           // conv1x1.w (256,256)
  AttnP dca[3]; FFP dff[3]; LNP dl1[3], dl2[3], dl3[3]; AttnP dsa[3];
  for (int l = 0; l < 3; ++l) {       // dec_layers: ca, ff, ln1, ln2, ln3, sa
    AttnP& a = dca[l]; a.bk = nxt(); a.bo = nxt(); a.bq = nxt(); a.bv = nxt();
                       a.wk = nxt(); a.wo = nxt(); a.wq = nxt(); a.wv = nxt();
    FFP& f = dff[l]; f.b1 = nxt(); f.b2 = nxt(); f.w1 = nxt(); f.w2 = nxt();
    dl1[l].b = nxt(); dl1[l].g = nxt();
    dl2[l].b = nxt(); dl2[l].g = nxt();
    dl3[l].b = nxt(); dl3[l].g = nxt();
    AttnP& s = dsa[l]; s.bk = nxt(); s.bo = nxt(); s.bq = nxt(); s.bv = nxt();
                       s.wk = nxt(); s.wo = nxt(); s.wq = nxt(); s.wv = nxt();
  }
  LNP dnorm; dnorm.b = nxt(); dnorm.g = nxt();
  ConvP ecv[3], eim[3];
  for (int l = 0; l < 3; ++l) { ecv[l].b = nxt(); ecv[l].beta = nxt(); ecv[l].g = nxt(); ecv[l].w = nxt(); }
  for (int l = 0; l < 3; ++l) { eim[l].b = nxt(); eim[l].beta = nxt(); eim[l].g = nxt(); eim[l].w = nxt(); }
  FFP eff[3]; LNP el1[3], el2[3]; AttnP esa[3];
  for (int l = 0; l < 3; ++l) {       // enc_layers: ff, ln1, ln2, sa
    FFP& f = eff[l]; f.b1 = nxt(); f.b2 = nxt(); f.w1 = nxt(); f.w2 = nxt();
    el1[l].b = nxt(); el1[l].g = nxt();
    el2[l].b = nxt(); el2[l].g = nxt();
    AttnP& s = esa[l]; s.bk = nxt(); s.bo = nxt(); s.bq = nxt(); s.bv = nxt();
                       s.wk = nxt(); s.wo = nxt(); s.wq = nxt(); s.wv = nxt();
  }
  LNP enorm; enorm.b = nxt(); enorm.g = nxt();
  const float* lin1_b = nxt(); const float* lin1_w = nxt();
  const float* lin2_b = nxt(); const float* lin2_w = nxt();
  const float* qpos   = nxt();

  // --- workspace carve-up (floats) ---
  float* wsf = (float*)d_ws;
  size_t off = 0;
  auto alloc = [&](size_t n) { float* p = wsf + off; off += (n + 3) & ~(size_t)3; return p; };
  float* bufA   = alloc(4 * 32 * 128 * 128);
  float* bufB   = alloc(4 * 64 * 64 * 64);
  float* img3   = alloc(4 * 128 * 32 * 32);
  float* cnv3   = alloc(4 * 128 * 32 * 32);
  float* bnm    = alloc(128);
  float* bnv    = alloc(128);
  float* tok    = alloc(4096 * 256);
  float* x      = alloc(4096 * 256);
  float* buf1   = alloc(4096 * 256);
  float* qb     = alloc(4096 * 256);
  float* kb     = alloc(4096 * 256);
  float* vb     = alloc(4096 * 256);
  float* attnb  = alloc(4096 * 256);
  float* ffmid  = alloc((size_t)4096 * 2048);
  float* scores = alloc((size_t)4 * 8 * 1024 * 1024);
  float* mem    = alloc(4096 * 256);
  float* y      = alloc(512 * 256);
  float* hs     = alloc(512 * 256);
  float* sp     = alloc(512 * 256);
  float* sp2    = alloc(512 * 16);
  float* recs   = alloc(512 * 208);

  // --- CNN encoders ---
  auto run_cnn = [&](const float* input, const ConvP* st3, float* out3) {
    conv3x3_kernel<<<(4 * 32 * 128 * 128 + 255) / 256, 256, 0, stream>>>(
        input, st3[0].w, st3[0].b, bufA, 4, 3, 32, 128, 128, 1, 128, 128);
    bn_stats_kernel<<<32, 256, 0, stream>>>(bufA, bnm, bnv, 4, 32, 16384);
    bn_apply_relu_kernel<<<(4 * 32 * 16384 + 255) / 256, 256, 0, stream>>>(
        bufA, st3[0].g, st3[0].beta, bnm, bnv, 4 * 32 * 16384, 32, 16384);
    conv3x3_kernel<<<(4 * 64 * 64 * 64 + 255) / 256, 256, 0, stream>>>(
        bufA, st3[1].w, st3[1].b, bufB, 4, 32, 64, 128, 128, 2, 64, 64);
    bn_stats_kernel<<<64, 256, 0, stream>>>(bufB, bnm, bnv, 4, 64, 4096);
    bn_apply_relu_kernel<<<(4 * 64 * 4096 + 255) / 256, 256, 0, stream>>>(
        bufB, st3[1].g, st3[1].beta, bnm, bnv, 4 * 64 * 4096, 64, 4096);
    conv3x3_kernel<<<(4 * 128 * 32 * 32 + 255) / 256, 256, 0, stream>>>(
        bufB, st3[2].w, st3[2].b, out3, 4, 64, 128, 64, 64, 2, 32, 32);
    bn_stats_kernel<<<128, 256, 0, stream>>>(out3, bnm, bnv, 4, 128, 1024);
    bn_apply_relu_kernel<<<(4 * 128 * 1024 + 255) / 256, 256, 0, stream>>>(
        out3, st3[2].g, st3[2].beta, bnm, bnv, 4 * 128 * 1024, 128, 1024);
  };
  run_cnn(img,    eim, img3);
  run_cnn(canvas, ecv, cnv3);

  build_tokens_kernel<<<4096, 256, 0, stream>>>(img3, cnv3, tok);
  linear(stream, tok, c1w, c1b, nullptr, x, 4096, 256, 256, 0);   // conv1x1 as GEMM

  // --- transformer encoder ---
  for (int l = 0; l < 3; ++l) {
    run_mha(stream, x, 1024, x, 1024, esa[l], x, qb, kb, vb, scores, attnb, buf1);
    layernorm_kernel<<<4096, 256, 0, stream>>>(buf1, el1[l].g, el1[l].b, x);
    linear(stream, x, eff[l].w1, eff[l].b1, nullptr, ffmid, 4096, 2048, 256, 1);
    linear(stream, ffmid, eff[l].w2, eff[l].b2, x, buf1, 4096, 256, 2048, 0);
    layernorm_kernel<<<4096, 256, 0, stream>>>(buf1, el2[l].g, el2[l].b, x);
  }
  layernorm_kernel<<<4096, 256, 0, stream>>>(x, enorm.g, enorm.b, mem);

  // --- transformer decoder ---
  bcast_kernel<<<(512 * 256 + 255) / 256, 256, 0, stream>>>(qpos, y, 512 * 256, 128 * 256);
  for (int l = 0; l < 3; ++l) {
    run_mha(stream, y, 128, y, 128, dsa[l], y, qb, kb, vb, scores, attnb, buf1);
    layernorm_kernel<<<512, 256, 0, stream>>>(buf1, dl1[l].g, dl1[l].b, y);
    run_mha(stream, y, 128, mem, 1024, dca[l], y, qb, kb, vb, scores, attnb, buf1);
    layernorm_kernel<<<512, 256, 0, stream>>>(buf1, dl2[l].g, dl2[l].b, y);
    linear(stream, y, dff[l].w1, dff[l].b1, nullptr, ffmid, 512, 2048, 256, 1);
    linear(stream, ffmid, dff[l].w2, dff[l].b2, y, buf1, 512, 256, 2048, 0);
    layernorm_kernel<<<512, 256, 0, stream>>>(buf1, dl3[l].g, dl3[l].b, y);
  }
  layernorm_kernel<<<512, 256, 0, stream>>>(y, dnorm.g, dnorm.b, hs);

  // --- heads ---
  linear(stream, hs, lin1_w, lin1_b, nullptr, sp, 512, 256, 256, 1);
  { // lin2: N=10, padded ldc=16; clamped loaders keep reads in-bounds
    dim3 grid(1, 4, 1);
    gemm_kernel<false><<<grid, 256, 0, stream>>>(sp, 256, 0, 0, lin2_w, 256, 0, 0, lin2_b,
                                                 nullptr, 0, 0, 0, sp2, 16, 0, 0,
                                                 512, 10, 256, 1, 1.0f, 0);
  }
  heads_kernel<<<2, 256, 0, stream>>>(sp2, recs);

  // --- rasterize ---
  rasterize_kernel<<<dim3(64, 4), 256, 0, stream>>>(recs, (float*)d_out);
}
